// ABS_IMPL_43482248904840
// MI455X (gfx1250) — matrix operand order repaired and checked by reading, not compiled, not run on hardware
//
#include <hip/hip_runtime.h>
#include <hip/hip_bf16.h>

// ---------------------------------------------------------------------------
// Sizes (compile-time constants from the reference)
// ---------------------------------------------------------------------------
#define BATCH   8
#define CDIM    256
#define NTOK    1568          // 8*14*14
#define HEADS   4
#define DHEAD   128
#define INNER   512           // HEADS*DHEAD
#define OQKV    1536          // 3*INNER
#define FD      8
#define HD      14
#define WD      14

typedef __attribute__((ext_vector_type(16))) _Float16 v16h;
typedef __attribute__((ext_vector_type(8)))  _Float16 v8h;
typedef __attribute__((ext_vector_type(8)))  float    v8f;
typedef __attribute__((ext_vector_type(4)))  int      v4i;

// ---------------------------------------------------------------------------
// Async global->LDS (gfx1250 GLOBAL_LOAD_ASYNC_TO_LDS_B128, ASYNCcnt-tracked).
// Probe result (round 2 diagnostic): builtin exists, param0 = v4i __device__*
// (addrspace(1)), so pass typed AS pointers. Fallback stays for portability.
// ---------------------------------------------------------------------------
#if defined(__has_builtin)
#if __has_builtin(__builtin_amdgcn_global_load_async_to_lds_b128)
#define HAVE_ASYNC_LDS 1
#endif
#endif

__device__ __forceinline__ void async_cp16(const _Float16* g, _Float16* l) {
#ifdef HAVE_ASYNC_LDS
  __builtin_amdgcn_global_load_async_to_lds_b128(
      (__attribute__((address_space(1))) v4i*)g,
      (__attribute__((address_space(3))) v4i*)l, 0, 0);
#else
  *(uint4*)l = *(const uint4*)g;
#endif
}

#ifdef HAVE_ASYNC_LDS
#if __has_builtin(__builtin_amdgcn_s_wait_asynccnt)
#define WAIT_ASYNC(n) __builtin_amdgcn_s_wait_asynccnt(n)
#else
#define WAIT_ASYNC(n) asm volatile("s_wait_asynccnt " #n ::: "memory")
#endif
#else
#define WAIT_ASYNC(n) ((void)0)
#endif

// ---------------------------------------------------------------------------
// Fragment helpers for V_WMMA_F32_16X16X32_F16 (ISA 7.12.2 layouts)
// A 16x32 f16: lane L (m=L&15, khalf=(L>>4)*8): elems[0..7]=A[m][khalf+i],
//              elems[8..15]=A[m][16+khalf+i]  -> two contiguous 16B loads.
// B 32x16 f16: lane L (n=L&15, khalf=(L>>4)*8): elems[0..7]=B[khalf+i][n],
//              elems[8..15]=B[16+khalf+i][n] (same k order as A); with
//              row-major [n][k] storage -> two 16B per-lane loads.
// C/D 16x16 f32: VGPR r holds row 8*(L>>4)+r, col L&15.
// ---------------------------------------------------------------------------
__device__ __forceinline__ v16h mk16(v8h lo, v8h hi) {
  v16h r;
#pragma unroll
  for (int i = 0; i < 8; ++i) { r[i] = lo[i]; r[8 + i] = hi[i]; }
  return r;
}
__device__ __forceinline__ v16h load_afrag(const _Float16* p0, const _Float16* p1) {
  return mk16(*(const v8h*)p0, *(const v8h*)p1);
}
__device__ __forceinline__ v16h load_bfrag(const _Float16* p) {
  int g8 = ((threadIdx.x & 31) >> 4) * 8;
  return mk16(*(const v8h*)(p - g8), *(const v8h*)(p + 16 - g8));
}
__device__ __forceinline__ v8f wmma_f16(v16h a, v16h b, v8f c) {
  return __builtin_amdgcn_wmma_f32_16x16x32_f16(false, a, false, b, (short)0, c,
                                                false, false);
}

// ---------------------------------------------------------------------------
// Kernel 0: pack fp32 -> f16 staging.
//   Xt[b][j][c] = fmap[b][c][j]   (transpose so GEMM A-fragments are contiguous)
//   Wh[o][c]    = W_qkv[o][c]
// ---------------------------------------------------------------------------
#define NXT (BATCH * NTOK * CDIM)
#define NWH (OQKV * CDIM)

__global__ __launch_bounds__(256) void pack_kernel(const float* __restrict__ fmap,
                                                   const float* __restrict__ Wq,
                                                   _Float16* __restrict__ Xt,
                                                   _Float16* __restrict__ Wh) {
  int idx = blockIdx.x * 256 + threadIdx.x;
  if (idx < NXT) {
    int b = idx / (NTOK * CDIM);
    int r = idx - b * (NTOK * CDIM);
    int j = r / CDIM;
    int c = r - j * CDIM;
    Xt[idx] = (_Float16)fmap[((size_t)b * CDIM + c) * NTOK + j];
  } else {
    int k = idx - NXT;  // grid sized exactly: k < NWH
    Wh[k] = (_Float16)Wq[k];
  }
}

// ---------------------------------------------------------------------------
// Kernel 1: QKV projection GEMM with WMMA.
//   out[j][o] = sum_c Xt[b][j][c] * Wh[o][c]      (per batch)
// One wave: 16 j-rows x 64 o-cols (4 output tiles, A reused), K=256 in 8 steps.
// Epilogue scatters to Qh (scaled), Kh (+positional emb), Vt (transposed).
// ---------------------------------------------------------------------------
#define JT   (NTOK / 16)     // 98
#define OG   (OQKV / 64)     // 24
#define QKV_WAVES (BATCH * JT * OG)

__global__ __launch_bounds__(256) void qkv_kernel(
    const _Float16* __restrict__ Xt, const _Float16* __restrict__ Wh,
    const float* __restrict__ pf, const float* __restrict__ ph,
    const float* __restrict__ pw, _Float16* __restrict__ Qh,
    _Float16* __restrict__ Kh, _Float16* __restrict__ Vt) {
  int wave = blockIdx.x * 8 + (threadIdx.x >> 5);
  int lane = threadIdx.x & 31;
  int b   = wave / (JT * OG);
  int rem = wave - b * (JT * OG);
  int jt  = rem / OG;
  int og  = rem - jt * OG;
  int jbase = jt * 16;
  int obase = og * 64;
  int mrow = lane & 15;
  int grp  = lane >> 4;

  const _Float16* xrow = Xt + ((size_t)b * NTOK + jbase + mrow) * CDIM;

  v8f zero = {0.f, 0.f, 0.f, 0.f, 0.f, 0.f, 0.f, 0.f};
  v8f acc[4];
#pragma unroll
  for (int ot = 0; ot < 4; ++ot) acc[ot] = zero;

#pragma unroll
  for (int kc = 0; kc < 8; ++kc) {
    int cb = kc * 32;
    v16h a = load_afrag(xrow + cb + grp * 8, xrow + cb + 16 + grp * 8);
#pragma unroll
    for (int ot = 0; ot < 4; ++ot) {
      int o = obase + ot * 16 + mrow;
      v16h bf = load_bfrag(Wh + (size_t)o * CDIM + cb + grp * 16);
      acc[ot] = wmma_f16(a, bf, acc[ot]);
    }
  }

  const float scale = 0.08838834764831845f;  // DHEAD^-0.5
#pragma unroll
  for (int ot = 0; ot < 4; ++ot) {
    int o    = obase + ot * 16 + mrow;
    int sect = o >> 9;        // 0=q 1=k 2=v (uniform per wave/ot: 64 | 512)
    int oo   = o & 511;
    int head = oo >> 7;
    int d    = oo & 127;
    size_t bh = (size_t)(b * HEADS + head);
#pragma unroll
    for (int r = 0; r < 8; ++r) {
      int j   = jbase + grp * 8 + r;
      float v = acc[ot][r];
      if (sect == 0) {
        Qh[(bh * NTOK + j) * DHEAD + d] = (_Float16)(v * scale);
      } else if (sect == 1) {
        int jf = j / (HD * WD);
        int jr = j - jf * (HD * WD);
        int jh = jr / WD;
        int jw = jr - jh * WD;
        float e = pf[jf * DHEAD + d] + ph[jh * DHEAD + d] + pw[jw * DHEAD + d];
        Kh[(bh * NTOK + j) * DHEAD + d] = (_Float16)(v + e);
      } else {
        Vt[(bh * DHEAD + d) * NTOK + j] = (_Float16)v;  // transposed for P@V
      }
    }
  }
}

// ---------------------------------------------------------------------------
// Kernel 2: cooperative flash attention.
// One block (8 waves) shares a (b, head); each wave owns a 16-query tile.
// 49 key-steps of 32: K tile (32x128, 8KB) and V tile (128x32, 8KB) are staged
// into LDS by the whole block via async global->LDS (double-buffered, DMA of
// step j+1 overlaps compute of step j, ASYNCcnt-pipelined), then each wave:
// 8 WMMA (QK'), online softmax (wave32 shfl_xor row reductions), P transposed
// through a per-wave LDS slab (s_wait_dscnt), 8 WMMA (P@V).
// ---------------------------------------------------------------------------
#define BPH 13                       // blocks per (b,h): 13*8 >= 98 q-tiles
#define ATT_BLOCKS (BATCH * HEADS * BPH)

__global__ __launch_bounds__(256) void attn_kernel(
    const _Float16* __restrict__ Qh, const _Float16* __restrict__ Kh,
    const _Float16* __restrict__ Vt, float* __restrict__ out) {
  __shared__ _Float16 Ktile[2][32][128];   // [buf][key][d]      2 x 8KB
  __shared__ _Float16 Vtile[2][128][32];   // [buf][d][key]      2 x 8KB
  __shared__ _Float16 plds[8][16][32];     // per-wave P staging      8KB

  int tid   = threadIdx.x;
  int wslot = tid >> 5;
  int lane  = tid & 31;
  int bhid  = blockIdx.x / BPH;            // b*HEADS + h
  int blk13 = blockIdx.x - bhid * BPH;
  int b = bhid >> 2;
  int h = bhid & 3;
  size_t bh = (size_t)bhid;

  int qt    = blk13 * 8 + wslot;
  bool valid = (qt < JT);                  // wave-uniform
  int qbase = qt * 16;
  int mrow  = lane & 15;
  int grp   = lane >> 4;

  // Resident Q A-fragments (K=128 -> 4 chunks of 32)
  v16h qa[4];
  if (valid) {
    const _Float16* qrow = Qh + (bh * NTOK + qbase + mrow) * DHEAD;
#pragma unroll
    for (int kc = 0; kc < 4; ++kc)
      qa[kc] = load_afrag(qrow + kc * 32 + grp * 8, qrow + kc * 32 + 16 + grp * 8);
  }

  v8f zero = {0.f, 0.f, 0.f, 0.f, 0.f, 0.f, 0.f, 0.f};
  v8f acc[8];
#pragma unroll
  for (int dt = 0; dt < 8; ++dt) acc[dt] = zero;
  float mrun[8], lrun[8];
#pragma unroll
  for (int r = 0; r < 8; ++r) { mrun[r] = -1e30f; lrun[r] = 0.f; }

  const float L2E = 1.4426950408889634f;
  const int NSTEP = NTOK / 32;  // 49

  // ---- stage helper: 256 threads move K tile (512x16B) + V tile (512x16B):
  //      4 async b128 issues per thread -> per-wave ASYNCcnt += 4 ----
  auto stage = [&](int buf, int jbase) {
#pragma unroll
    for (int u = 0; u < 2; ++u) {
      int c  = tid + u * 256;              // 0..511
      int kr = c >> 4, k8 = c & 15;        // K: 32 rows x 16 chunks
      async_cp16(Kh + (bh * NTOK + jbase + kr) * DHEAD + k8 * 8,
                 &Ktile[buf][kr][k8 * 8]);
      int vr = c >> 2, v8_ = c & 3;        // V: 128 rows x 4 chunks
      async_cp16(Vt + (bh * DHEAD + vr) * NTOK + jbase + v8_ * 8,
                 &Vtile[buf][vr][v8_ * 8]);
    }
  };

  stage(0, 0);  // prologue: buffer 0 <- keys [0,32)

  for (int jb = 0; jb < NSTEP; ++jb) {
    int cur = jb & 1;
    if (jb + 1 < NSTEP) {
      stage(cur ^ 1, (jb + 1) * 32);   // overlap DMA of next tile
      WAIT_ASYNC(4);                   // oldest 4 (current tile) complete
    } else {
      WAIT_ASYNC(0);
    }
    __syncthreads();                   // current tile visible block-wide

    if (valid) {
      // ---- S = Q * K'^T (two 16x16 tiles covering 32 keys) ----
      v8f s[2];
      s[0] = zero; s[1] = zero;
#pragma unroll
      for (int t = 0; t < 2; ++t) {
#pragma unroll
        for (int kc = 0; kc < 4; ++kc) {
          v16h kb = load_bfrag(&Ktile[cur][t * 16 + mrow][kc * 32 + grp * 16]);
          s[t] = wmma_f16(qa[kc], kb, s[t]);
        }
      }

      // ---- online softmax (rows on VGPR index r, cols across 16 lanes) ----
      float p0[8], p1[8];
#pragma unroll
      for (int r = 0; r < 8; ++r) {
        float mx = fmaxf(s[0][r], s[1][r]);
#pragma unroll
        for (int off = 1; off < 16; off <<= 1)
          mx = fmaxf(mx, __shfl_xor(mx, off, 32));
        float mn    = fmaxf(mrun[r], mx);
        float alpha = exp2f((mrun[r] - mn) * L2E);
        float a0    = exp2f((s[0][r] - mn) * L2E);
        float a1    = exp2f((s[1][r] - mn) * L2E);
        float rs = a0 + a1;
#pragma unroll
        for (int off = 1; off < 16; off <<= 1) rs += __shfl_xor(rs, off, 32);
        lrun[r] = lrun[r] * alpha + rs;
        mrun[r] = mn;
        p0[r] = a0; p1[r] = a1;
#pragma unroll
        for (int dt = 0; dt < 8; ++dt) acc[dt][r] *= alpha;
      }

      // ---- transpose P (C layout -> A layout) through per-wave LDS ----
#pragma unroll
      for (int r = 0; r < 8; ++r) {
        plds[wslot][grp * 8 + r][mrow]      = (_Float16)p0[r];
        plds[wslot][grp * 8 + r][16 + mrow] = (_Float16)p1[r];
      }
      asm volatile("s_wait_dscnt 0" ::: "memory");  // same-wave LDS RAW
      v16h pa = load_afrag(&plds[wslot][mrow][grp * 8],
                           &plds[wslot][mrow][16 + grp * 8]);

      // ---- O += P(16x32) * V(32x128) : 8 d-tiles ----
#pragma unroll
      for (int dt = 0; dt < 8; ++dt) {
        v16h vb = load_bfrag(&Vtile[cur][dt * 16 + mrow][grp * 16]);
        acc[dt] = wmma_f16(pa, vb, acc[dt]);
      }
    }

    __syncthreads();  // all readers done before next DMA overwrites this buffer
  }

  // ---- epilogue: normalize and scatter to out[b][(h d)][n] ----
  if (valid) {
#pragma unroll
    for (int dt = 0; dt < 8; ++dt) {
      int ocol = h * DHEAD + dt * 16 + mrow;
#pragma unroll
      for (int r = 0; r < 8; ++r) {
        out[((size_t)b * INNER + ocol) * NTOK + qbase + grp * 8 + r] =
            acc[dt][r] / lrun[r];
      }
    }
  }
}

// ---------------------------------------------------------------------------
// Host launcher
// ---------------------------------------------------------------------------
extern "C" void kernel_launch(void* const* d_in, const int* in_sizes, int n_in,
                              void* d_out, int out_size, void* d_ws, size_t ws_size,
                              hipStream_t stream) {
  const float* fmap = (const float*)d_in[0];
  const float* Wq   = (const float*)d_in[1];
  const float* pf   = (const float*)d_in[2];
  const float* ph   = (const float*)d_in[3];
  const float* pw   = (const float*)d_in[4];
  float* out        = (float*)d_out;

  char* ws = (char*)d_ws;
  const size_t XT_B = (size_t)NXT * 2;                           //  6,422,528
  const size_t WH_B = (size_t)NWH * 2;                           //    786,432
  const size_t QH_B = (size_t)BATCH * HEADS * NTOK * DHEAD * 2;  // 12,845,056
  _Float16* Xt = (_Float16*)(ws);
  _Float16* Wh = (_Float16*)(ws + XT_B);
  _Float16* Qh = (_Float16*)(ws + XT_B + WH_B);
  _Float16* Kh = (_Float16*)(ws + XT_B + WH_B + QH_B);
  _Float16* Vt = (_Float16*)(ws + XT_B + WH_B + 2 * QH_B);
  (void)ws_size; (void)in_sizes; (void)n_in; (void)out_size;

  // Pack: exact grid (3,604,480 / 256 = 14,080)
  pack_kernel<<<(NXT + NWH) / 256, 256, 0, stream>>>(fmap, Wq, Xt, Wh);

  // QKV GEMM: 18,816 waves -> 2,352 blocks of 8 waves (exact, EXEC all-ones)
  qkv_kernel<<<QKV_WAVES / 8, 256, 0, stream>>>(Xt, Wh, pf, ph, pw, Qh, Kh, Vt);

  // Flash attention: 13 blocks per (b,h) -> 416 blocks of 8 waves
  attn_kernel<<<ATT_BLOCKS, 256, 0, stream>>>(Qh, Kh, Vt, out);
}
